// LIIFParametric3DConv_6296422055942
// MI455X (gfx1250) — matrix operand order repaired and checked by reading, not compiled, not run on hardware
//
#include <hip/hip_runtime.h>
#include <stdint.h>

typedef __attribute__((ext_vector_type(16))) _Float16 v16h;
typedef __attribute__((ext_vector_type(8)))  float    v8f;

// ---- fixed problem shapes (from setup_inputs) ----
#define BB   2
#define TT   32
#define HH   96
#define WW   96
#define QQ   32
#define CC   32
#define HID  64
#define HWP  (HH*WW)            // 9216
#define NTILE (HWP/16)          // 576 16-pixel tiles
#define NUNIT (BB*QQ*NTILE)     // 36864 wave-units

// ---- workspace layout (bytes) ----
#define XENC_BYTES  37748736u   // B*T*HW*C halfs
#define BIAS0_OFF   XENC_BYTES                  // [B][Q][2][64] f32
#define IDX_OFF     (BIAS0_OFF + 32768u)        // [B][Q][2] int
#define WGT_OFF     (IDX_OFF   + 512u)          // [B][Q][2] f32
#define B0_OFF      (WGT_OFF   + 512u)          // [4][32][8] u32
#define B1_OFF      (B0_OFF    + 4096u)         // [2][4][32][8] u32
#define B2_OFF      (B1_OFF    + 8192u)         // [2][32][8] u32

#define INV2PI 0.15915494309189535f
#define W0SIREN 30.0f

union Frag16 { v16h v; uint32_t u[8]; };

__device__ __forceinline__ void ld16(uint32_t* d, const void* s) {
  uint4 t = *reinterpret_cast<const uint4*>(s);
  d[0] = t.x; d[1] = t.y; d[2] = t.z; d[3] = t.w;
}

__device__ __forceinline__ uint32_t pack_h2(float a, float b) {
  union { _Float16 h[2]; uint32_t u; } x;
  x.h[0] = (_Float16)a; x.h[1] = (_Float16)b;
  return x.u;
}

__device__ __forceinline__ void branch_calc(float t, float v, int* idx, float* rel) {
  float tc = t + (1.0f / (TT - 1)) * v + 1e-6f;
  tc = fminf(fmaxf(tc, -1.0f + 1e-6f), 1.0f - 1e-6f);
  float pos = (tc + 1.0f) * 0.5f * (float)(TT - 1);
  int id = (int)rintf(pos);
  id = min(max(id, 0), TT - 1);
  float qc = -1.0f + 2.0f * (float)id / (float)(TT - 1);
  *idx = id;
  *rel = (t - qc) * (float)(TT - 1);
}

// ---------------------------------------------------------------------------
// Kernel 1: 3x3x3 conv (2->32 ch), SAME pad, output f16 in [B][T][HW][C]
// ---------------------------------------------------------------------------
__global__ __launch_bounds__(256) void conv3d_kernel(
    const float* __restrict__ xr, const float* __restrict__ xi,
    const float* __restrict__ cw, const float* __restrict__ cb,
    _Float16* __restrict__ xenc) {
  __shared__ float wl[CC * 54];
  __shared__ float bl[CC];
  int tid = threadIdx.x;
  for (int i = tid; i < CC * 54; i += 256) wl[i] = cw[i];
  if (tid < CC) bl[tid] = cb[tid];
  __syncthreads();

  int vox = blockIdx.x * 256 + tid;          // grid is exact: B*T*H*W/256
  int w = vox % WW;
  int h = (vox / WW) % HH;
  int t = (vox / (WW * HH)) % TT;
  int b = vox / (WW * HH * TT);

  float acc[CC];
#pragma unroll
  for (int c = 0; c < CC; ++c) acc[c] = bl[c];

  for (int dt = -1; dt <= 1; ++dt) {
    int tt = t + dt;
    bool tok = (unsigned)tt < (unsigned)TT;
    for (int dh = -1; dh <= 1; ++dh) {
      int hh = h + dh;
      bool hok = (unsigned)hh < (unsigned)HH;
      for (int dw = -1; dw <= 1; ++dw) {
        int ww2 = w + dw;
        bool wok = (unsigned)ww2 < (unsigned)WW;
        float vr = 0.0f, vi = 0.0f;
        if (tok && hok && wok) {
          size_t off = ((size_t)(b * TT + tt) * HH + hh) * WW + ww2;
          vr = xr[off];
          vi = xi[off];
        }
        int tap = ((dt + 1) * 3 + (dh + 1)) * 3 + (dw + 1);
#pragma unroll
        for (int c = 0; c < CC; ++c)
          acc[c] += vr * wl[c * 54 + tap] + vi * wl[c * 54 + 27 + tap];
      }
    }
  }

  uint32_t pk[16];
#pragma unroll
  for (int i = 0; i < 16; ++i) pk[i] = pack_h2(acc[2 * i], acc[2 * i + 1]);
  uint4* op = reinterpret_cast<uint4*>(xenc + (size_t)vox * CC);
#pragma unroll
  for (int i = 0; i < 4; ++i) {
    uint4 v;
    v.x = pk[4 * i]; v.y = pk[4 * i + 1]; v.z = pk[4 * i + 2]; v.w = pk[4 * i + 3];
    op[i] = v;
  }
}

// ---------------------------------------------------------------------------
// Kernel 2: per-(b,q,branch) gather index / ensemble weights, folded layer-0
// bias (rel*w0[32,:]+b0), and WMMA-B-layout weight packs for w0/w1/w2.
// ---------------------------------------------------------------------------
__global__ __launch_bounds__(256) void prep_kernel(
    const float* __restrict__ t_coord, const float* __restrict__ w0,
    const float* __restrict__ b0, const float* __restrict__ w1,
    const float* __restrict__ w2, float* __restrict__ bias0,
    int* __restrict__ idxb, float* __restrict__ wgtb,
    uint32_t* __restrict__ B0p, uint32_t* __restrict__ B1p,
    uint32_t* __restrict__ B2p) {
  int tid = threadIdx.x;

  // per-(b,q): indices + ensemble weights
  for (int i = tid; i < BB * QQ; i += 256) {
    float t = t_coord[i];
    int im, ip; float rm, rp;
    branch_calc(t, -1.0f, &im, &rm);
    branch_calc(t, 1.0f, &ip, &rp);
    float dm = fabsf(rm), dp = fabsf(rp);
    float tot = dm + dp;
    idxb[i * 2 + 0] = im;
    idxb[i * 2 + 1] = ip;
    wgtb[i * 2 + 0] = dp / tot;   // weight applied to minus-branch pred
    wgtb[i * 2 + 1] = dm / tot;   // weight applied to plus-branch pred
  }

  // folded layer-0 bias: bias0[b][q][br][j] = rel*w0[32][j] + b0[j]
  for (int i = tid; i < BB * QQ * 2 * HID; i += 256) {
    int j = i & 63;
    int br = (i >> 6) & 1;
    int bq = i >> 7;
    float t = t_coord[bq];
    int id; float rel;
    branch_calc(t, br ? 1.0f : -1.0f, &id, &rel);
    bias0[i] = rel * w0[32 * HID + j] + b0[j];
  }

  // B0 pack: [nt(4)][lane(32)][j(8)]; lane<16 -> col=nt*16+lane, K=0..7,16..23
  //          lane>=16 -> col=nt*16+(lane-16), K=8..15,24..31 (pairs per dword)
  for (int i = tid; i < 4 * 32 * 8; i += 256) {
    int j = i & 7;
    int lane = (i >> 3) & 31;
    int nt = i >> 8;
    int col = nt * 16 + (lane & 15);
    int k = ((lane < 16) ? 0 : 8) + ((j < 4) ? 0 : 8) + 2 * j;
    B0p[i] = pack_h2(w0[k * HID + col], w0[(k + 1) * HID + col]);
  }

  // B1 pack: [kh(2)][nt(4)][lane(32)][j(8)] over w1 (64x64)
  for (int i = tid; i < 2 * 4 * 32 * 8; i += 256) {
    int j = i & 7;
    int lane = (i >> 3) & 31;
    int nt = (i >> 8) & 3;
    int kh = i >> 10;
    int col = nt * 16 + (lane & 15);
    int k = kh * 32 + ((lane < 16) ? 0 : 8) + ((j < 4) ? 0 : 8) + 2 * j;
    B1p[i] = pack_h2(w1[k * HID + col], w1[(k + 1) * HID + col]);
  }

  // B2 pack: [kh(2)][lane(32)][j(8)] over w2 (64x1); only column 0 non-zero
  for (int i = tid; i < 2 * 32 * 8; i += 256) {
    int j = i & 7;
    int lane = (i >> 3) & 31;
    int kh = i >> 8;
    uint32_t val = 0u;
    if ((lane & 15) == 0) {
      int k = kh * 32 + ((lane < 16) ? 0 : 8) + ((j < 4) ? 0 : 8) + 2 * j;
      val = pack_h2(w2[k], w2[k + 1]);
    }
    B2p[i] = val;
  }
}

// ---------------------------------------------------------------------------
// Kernel 3: fused SIREN + local ensemble. One wave32 per (b,q,16-pixel tile).
// ---------------------------------------------------------------------------
__device__ __forceinline__ Frag16 load_bfrag(const uint32_t* base, int mat, int lane) {
  Frag16 f;
  const uint32_t* p = base + ((mat * 32 + lane) << 3);
  ld16(f.u, p);
  ld16(f.u + 4, p + 4);
  return f;
}

__global__ __launch_bounds__(128) void siren_kernel(
    const _Float16* __restrict__ xenc, const float* __restrict__ bias0,
    const int* __restrict__ idxb, const float* __restrict__ wgtb,
    const uint32_t* __restrict__ B0p, const uint32_t* __restrict__ B1p,
    const uint32_t* __restrict__ B2p, const float* __restrict__ b1,
    const float* __restrict__ b2, float* __restrict__ out) {
  __shared__ __align__(16) _Float16 act[4][16 * HID];   // per-wave 2KB staging

  const int lane = threadIdx.x & 31;
  const int wv = threadIdx.x >> 5;
  const int unit = blockIdx.x * 4 + wv;           // exact: NUNIT units
  const int tile = unit % NTILE;
  const int bq = unit / NTILE;                    // b*QQ + q
  const int pixbase = tile * 16;

  const int rm = lane & 15;                       // row (A) / column (B,C,D)
  const int hiHalf = lane >> 4;

  _Float16* myact = &act[wv][0];

  float b1v[4];
#pragma unroll
  for (int nt = 0; nt < 4; ++nt) b1v[nt] = b1[nt * 16 + rm];

  v8f predC = {};                                 // layer-2 accumulator (both branches)

  for (int br = 0; br < 2; ++br) {
    const int tIdx = idxb[bq * 2 + br];
    const float wgt = wgtb[bq * 2 + br];
    const float* bias0p = bias0 + (bq * 2 + br) * HID;

    // ---- layer 0 A fragment straight from global (f16, [pixel][32ch]) ----
    const _Float16* arow =
        xenc + (((size_t)(tIdx) + (size_t)(bq >> 5) * TT) * HWP + (pixbase + rm)) * CC;
    const int c0 = hiHalf ? 8 : 0;
    Frag16 a0;
    ld16(a0.u, arow + c0);
    ld16(a0.u + 4, arow + c0 + 16);
    __builtin_prefetch((const void*)(arow + 16 * CC), 0, 3);   // next tile

    // ---- layer 0: K=32 GEMM, 4 N-tiles, bias folded into C ----
#pragma unroll
    for (int nt = 0; nt < 4; ++nt) {
      Frag16 bf = load_bfrag(B0p, nt, lane);
      float bv = bias0p[nt * 16 + rm];
      v8f c;
#pragma unroll
      for (int i = 0; i < 8; ++i) c[i] = bv;
      v8f d = __builtin_amdgcn_wmma_f32_16x16x32_f16(false, a0.v, false, bf.v,
                                                     (short)0, c, false, false);
#pragma unroll
      for (int r = 0; r < 8; ++r) {
        float s = __builtin_amdgcn_sinf(d[r] * (W0SIREN * INV2PI));
        myact[(r + hiHalf * 8) * HID + nt * 16 + rm] = (_Float16)s;
      }
    }
    asm volatile("s_wait_dscnt 0x0" ::: "memory");

    // ---- load layer-1 A fragments (both K halves) from LDS ----
    Frag16 a1[2];
#pragma unroll
    for (int kh = 0; kh < 2; ++kh) {
      int cb = kh * 32 + (hiHalf ? 8 : 0);
      ld16(a1[kh].u, &myact[rm * HID + cb]);
      ld16(a1[kh].u + 4, &myact[rm * HID + cb + 16]);
    }

    // ---- layer 1: K=64 (two chained WMMAs) per N-tile; sin; scale by wgt ----
#pragma unroll
    for (int nt = 0; nt < 4; ++nt) {
      v8f c;
#pragma unroll
      for (int i = 0; i < 8; ++i) c[i] = b1v[nt];
      Frag16 bk0 = load_bfrag(B1p, 0 * 4 + nt, lane);
      v8f d = __builtin_amdgcn_wmma_f32_16x16x32_f16(false, a1[0].v, false, bk0.v,
                                                     (short)0, c, false, false);
      Frag16 bk1 = load_bfrag(B1p, 1 * 4 + nt, lane);
      d = __builtin_amdgcn_wmma_f32_16x16x32_f16(false, a1[1].v, false, bk1.v,
                                                 (short)0, d, false, false);
#pragma unroll
      for (int r = 0; r < 8; ++r) {
        float s = __builtin_amdgcn_sinf(d[r] * (W0SIREN * INV2PI)) * wgt;
        myact[(r + hiHalf * 8) * HID + nt * 16 + rm] = (_Float16)s;
      }
    }
    asm volatile("s_wait_dscnt 0x0" ::: "memory");

    // ---- layer 2: head (w2 in column 0), accumulate across branches ----
    Frag16 a2[2];
#pragma unroll
    for (int kh = 0; kh < 2; ++kh) {
      int cb = kh * 32 + (hiHalf ? 8 : 0);
      ld16(a2[kh].u, &myact[rm * HID + cb]);
      ld16(a2[kh].u + 4, &myact[rm * HID + cb + 16]);
    }
    Frag16 c2k0 = load_bfrag(B2p, 0, lane);
    predC = __builtin_amdgcn_wmma_f32_16x16x32_f16(false, a2[0].v, false, c2k0.v,
                                                   (short)0, predC, false, false);
    Frag16 c2k1 = load_bfrag(B2p, 1, lane);
    predC = __builtin_amdgcn_wmma_f32_16x16x32_f16(false, a2[1].v, false, c2k1.v,
                                                   (short)0, predC, false, false);
  }

  // column 0 of D holds pred for the 16 pixels: lane 0 rows 0-7, lane 16 rows 8-15
  float b2v = b2[0];
  if (rm == 0) {
    float* op = out + (size_t)bq * HWP + pixbase + hiHalf * 8;
    float4 o0, o1;
    o0.x = predC[0] + b2v; o0.y = predC[1] + b2v;
    o0.z = predC[2] + b2v; o0.w = predC[3] + b2v;
    o1.x = predC[4] + b2v; o1.y = predC[5] + b2v;
    o1.z = predC[6] + b2v; o1.w = predC[7] + b2v;
    *reinterpret_cast<float4*>(op) = o0;
    *reinterpret_cast<float4*>(op + 4) = o1;
  }
}

// ---------------------------------------------------------------------------
extern "C" void kernel_launch(void* const* d_in, const int* in_sizes, int n_in,
                              void* d_out, int out_size, void* d_ws, size_t ws_size,
                              hipStream_t stream) {
  (void)in_sizes; (void)n_in; (void)out_size; (void)ws_size;
  const float* xr = (const float*)d_in[0];
  const float* xi = (const float*)d_in[1];
  const float* tc = (const float*)d_in[2];
  const float* cw = (const float*)d_in[3];
  const float* cb = (const float*)d_in[4];
  const float* w0 = (const float*)d_in[5];
  const float* b0 = (const float*)d_in[6];
  const float* w1 = (const float*)d_in[7];
  const float* b1 = (const float*)d_in[8];
  const float* w2 = (const float*)d_in[9];
  const float* b2 = (const float*)d_in[10];
  float* out = (float*)d_out;

  char* ws = (char*)d_ws;
  _Float16* xenc = (_Float16*)(ws);
  float* bias0 = (float*)(ws + BIAS0_OFF);
  int* idxb = (int*)(ws + IDX_OFF);
  float* wgtb = (float*)(ws + WGT_OFF);
  uint32_t* B0p = (uint32_t*)(ws + B0_OFF);
  uint32_t* B1p = (uint32_t*)(ws + B1_OFF);
  uint32_t* B2p = (uint32_t*)(ws + B2_OFF);

  conv3d_kernel<<<(BB * TT * HH * WW) / 256, 256, 0, stream>>>(xr, xi, cw, cb, xenc);
  prep_kernel<<<1, 256, 0, stream>>>(tc, w0, b0, w1, w2, bias0, idxb, wgtb, B0p, B1p, B2p);
  siren_kernel<<<NUNIT / 4, 128, 0, stream>>>(xenc, bias0, idxb, wgtb, B0p, B1p, B2p,
                                              b1, b2, out);
}
